// DiffeomorphicAttention_49039936586390
// MI455X (gfx1250) — compile-verified, matrix operand order checked by reading, not run on hardware
//
#include <hip/hip_runtime.h>

typedef _Float16 f16;
typedef _Float16 v16h __attribute__((ext_vector_type(16)));
typedef _Float16 h8   __attribute__((ext_vector_type(8)));
typedef _Float16 h4   __attribute__((ext_vector_type(4)));
typedef float    v8f  __attribute__((ext_vector_type(8)));

// workspace layout (bytes)
#define WS_QKVW_OFF   0u          // 24*4 frags * 512 halfs = 98304 B
#define WS_PROJW_OFF  98304u      // 8*4 frags  * 512 halfs = 32768 B
#define WS_BM_OFF     131072u     // 64*4*49*49 f32 = 2458624 B

// ---------------------------------------------------------------------------
// A/B fragment loader for v_wmma_f32_16x16x32_f16 (wave32).
// lane = {hi,m}: m = lane&15 (row / B-column), hi = lane>>4.
//   regs 0..7  : K = hi*8 + 0..7
//   regs 8..15 : K = 16 + hi*8 + 0..7
// Both 8-half groups are 16B contiguous and 16B aligned -> 2x ds_load_b128.
// ---------------------------------------------------------------------------
__device__ __forceinline__ v16h load_frag(const f16* base, int stride, int lid)
{
    const int m  = lid & 15;
    const int hi = lid >> 4;
    const f16* p = base + m * stride + hi * 8;
    h8 lo = *(const h8*)(p);
    h8 hh = *(const h8*)(p + 16);
    v16h a;
#pragma unroll
    for (int v = 0; v < 8; ++v) { a[v] = lo[v]; a[8 + v] = hh[v]; }
    return a;
}

// ---------------------------------------------------------------------------
// Setup 1: convert qkv_w [128,384] and proj_w [128,128] to f16, pre-swizzled
// into per-fragment, per-lane contiguous B-operand order.
// ---------------------------------------------------------------------------
__global__ void k_swizzle(const float* __restrict__ qkv_w,
                          const float* __restrict__ proj_w,
                          f16* __restrict__ wqkv, f16* __restrict__ wproj)
{
    int idx = blockIdx.x * blockDim.x + threadIdx.x;
    if (idx >= 49152 + 16384) return;
    int i   = (idx < 49152) ? idx : idx - 49152;
    int e    = i & 15;
    int lane = (i >> 4) & 31;
    int f    = i >> 9;
    int ks   = f & 3;
    int nt   = f >> 2;
    int v  = e >> 1, j = e & 1, hi = lane >> 4;
    int kl = (v < 4 ? 2*v : 16 + 2*(v - 4)) + hi * 8 + j;
    int k  = ks * 32 + kl;
    int n  = nt * 16 + (lane & 15);
    if (idx < 49152) wqkv[i]  = (f16)qkv_w[k * 384 + n];
    else             wproj[i] = (f16)proj_w[k * 128 + n];
}

// ---------------------------------------------------------------------------
// Setup 2: bias+mask table [64][4][49][49].
// bias(h,i,j) = MLP(rel(i,j)) with exact-erf GELU; table[w][h] = bias[h] + mask[w].
// ---------------------------------------------------------------------------
__global__ void k_biasmask(const float* __restrict__ mask,
                           const float* __restrict__ w1, const float* __restrict__ b1,
                           const float* __restrict__ w2, const float* __restrict__ b2,
                           float* __restrict__ bmt)
{
    int idx = blockIdx.x * blockDim.x + threadIdx.x;
    if (idx >= 64 * 2401) return;
    int w = idx / 2401, r = idx % 2401;
    int i = r / 49, j = r % 49;
    float r0 = (float)((i / 7) - (j / 7)) * (1.0f / 6.0f);
    float r1 = (float)((i % 7) - (j % 7)) * (1.0f / 6.0f);
    float a0 = b2[0], a1 = b2[1], a2 = b2[2], a3 = b2[3];
    for (int t = 0; t < 64; ++t) {
        float hp = r0 * w1[t] + r1 * w1[64 + t] + b1[t];
        float g  = 0.5f * hp * (1.0f + erff(hp * 0.70710678118654752f));
        a0 += g * w2[t * 4 + 0]; a1 += g * w2[t * 4 + 1];
        a2 += g * w2[t * 4 + 2]; a3 += g * w2[t * 4 + 3];
    }
    float mv = mask[w * 2401 + r];
    bmt[(w * 4 + 0) * 2401 + r] = a0 + mv;
    bmt[(w * 4 + 1) * 2401 + r] = a1 + mv;
    bmt[(w * 4 + 2) * 2401 + r] = a2 + mv;
    bmt[(w * 4 + 3) * 2401 + r] = a3 + mv;
}

// ---------------------------------------------------------------------------
// Main: one window (49x128) per 256-thread block (8 wave32).
// All 4 heads processed concurrently; 5 barriers total.
// ---------------------------------------------------------------------------
__global__ __launch_bounds__(256) void k_main(
    const float* __restrict__ x, const float* __restrict__ qkvb,
    const float* __restrict__ projb,
    const f16* __restrict__ wqkv, const f16* __restrict__ wproj,
    const float* __restrict__ bmt, float* __restrict__ out)
{
    __shared__ f16   sh_qk[64 * 256];   // q (cols 0..127, pre-scaled) | k (cols 128..255)
    __shared__ f16   sh_vT[128 * 64];   // v column-major: [c][token], c = h*32+d
    __shared__ f16   sh_sp[4 * 64 * 64];// scores, then (in-place) unnormalized probs
    __shared__ f16   sh_x [64 * 128];   // x staging; later head-concat attention output
    __shared__ float sh_inv[4 * 64];    // per (head,row) 1/sum

    const int b   = blockIdx.x;
    const int tid = threadIdx.x;
    const int lid = tid & 31;
    const int wv  = tid >> 5;
    const int ln  = lid & 15;
    const int hi  = lid >> 4;
    const float scale = 0.17677669529663687f;   // 32^-0.5

    // ---- stage x -> f16, zero-pad rows 49..63 (b128 global, b64 LDS) ----
    {
        const float4* xb4 = (const float4*)(x + (size_t)b * 49 * 128);
        for (int idx = tid; idx < 64 * 32; idx += 256) {
            int row = idx >> 5;
            float4 v4 = (row < 49) ? xb4[idx] : float4{0.f, 0.f, 0.f, 0.f};
            h4 hv = { (f16)v4.x, (f16)v4.y, (f16)v4.z, (f16)v4.w };
            *(h4*)(sh_x + idx * 4) = hv;
        }
    }
    __syncthreads();

    // ---- QKV GEMM: [64x128] x [128x384]; q/k/v epilogues split (uniform) ----
    // Q tiles (ntile 0..7), pre-scaled
#pragma unroll
    for (int i = 0; i < 4; ++i) {
        int t = wv + i * 8;
        int nt = t & 7, mt = t >> 3;
        v8f c = {};
#pragma unroll
        for (int ks = 0; ks < 4; ++ks) {
            v16h a  = load_frag(sh_x + mt * 16 * 128 + ks * 32, 128, lid);
            v16h bf = *(const v16h*)(wqkv + (size_t)((nt * 4 + ks) * 32 + lid) * 16);
            c = __builtin_amdgcn_wmma_f32_16x16x32_f16(false, a, false, bf,
                                                       (short)0, c, false, false);
        }
        int col = nt * 16 + ln;
        float bias = qkvb[col];
        f16* dst = sh_qk + (mt * 16 + hi * 8) * 256 + col;
#pragma unroll
        for (int r = 0; r < 8; ++r) dst[r * 256] = (f16)((c[r] + bias) * scale);
    }
    // K tiles (ntile 8..15)
#pragma unroll
    for (int i = 0; i < 4; ++i) {
        int t = wv + i * 8;
        int nt = t & 7, mt = t >> 3;
        v8f c = {};
#pragma unroll
        for (int ks = 0; ks < 4; ++ks) {
            v16h a  = load_frag(sh_x + mt * 16 * 128 + ks * 32, 128, lid);
            v16h bf = *(const v16h*)(wqkv + (size_t)(((nt + 8) * 4 + ks) * 32 + lid) * 16);
            c = __builtin_amdgcn_wmma_f32_16x16x32_f16(false, a, false, bf,
                                                       (short)0, c, false, false);
        }
        int col = nt * 16 + ln;
        float bias = qkvb[128 + col];
        f16* dst = sh_qk + (mt * 16 + hi * 8) * 256 + 128 + col;
#pragma unroll
        for (int r = 0; r < 8; ++r) dst[r * 256] = (f16)(c[r] + bias);
    }
    // V tiles (ntile 16..23), stored transposed (column-major)
#pragma unroll
    for (int i = 0; i < 4; ++i) {
        int t = wv + i * 8;
        int nt = t & 7, mt = t >> 3;
        v8f c = {};
#pragma unroll
        for (int ks = 0; ks < 4; ++ks) {
            v16h a  = load_frag(sh_x + mt * 16 * 128 + ks * 32, 128, lid);
            v16h bf = *(const v16h*)(wqkv + (size_t)(((nt + 16) * 4 + ks) * 32 + lid) * 16);
            c = __builtin_amdgcn_wmma_f32_16x16x32_f16(false, a, false, bf,
                                                       (short)0, c, false, false);
        }
        int cc = nt * 16 + ln;                       // 0..127
        float bias = qkvb[256 + cc];
        f16* dst = sh_vT + cc * 64 + mt * 16 + hi * 8;
#pragma unroll
        for (int r = 0; r < 8; ++r) dst[r] = (f16)(c[r] + bias);
    }
    __syncthreads();

    // ---- S = q k^T, all heads: 64 tiles, 8 per wave (K = hd = 32) ----
#pragma unroll
    for (int i = 0; i < 8; ++i) {
        int t  = wv * 8 + i;        // q-frag (h,mt) constant over groups of 4 -> CSE
        int h  = t >> 4;
        int mt = (t >> 2) & 3;
        int nt = t & 3;
        v16h a  = load_frag(sh_qk + mt * 16 * 256 + h * 32, 256, lid);
        v16h bf = load_frag(sh_qk + nt * 16 * 256 + 128 + h * 32, 256, lid);
        v8f c = {};
        c = __builtin_amdgcn_wmma_f32_16x16x32_f16(false, a, false, bf,
                                                   (short)0, c, false, false);
        f16* dst = sh_sp + (h * 64 + mt * 16 + hi * 8) * 64 + nt * 16 + ln;
#pragma unroll
        for (int r = 0; r < 8; ++r) dst[r * 64] = (f16)c[r];
    }
    __syncthreads();

    // ---- softmax: one (head,row) per thread; exp stored in place (f16),
    //      1/sum deferred to the PV epilogue via sh_inv ----
    if (tid < 196) {
        int h = tid / 49;
        int i = tid - h * 49;
        f16* srow = sh_sp + (h * 64 + i) * 64;
        const float* bmr = bmt + (((size_t)(b & 63) * 4 + h) * 2401) + i * 49;
        float mx = -1e30f;
        for (int j = 0; j < 49; ++j)
            mx = fmaxf(mx, (float)srow[j] + bmr[j]);
        float s = 0.0f;
        for (int j = 0; j < 49; ++j) {
            float e = __expf((float)srow[j] + bmr[j] - mx);
            srow[j] = (f16)e;
            s += e;
        }
        for (int j = 49; j < 64; ++j) srow[j] = (f16)0.0f;
        sh_inv[h * 64 + i] = 1.0f / s;
    } else {
        int z = tid - 196;                  // 60 threads cover 4 heads x 15 pad rows
        int h = z / 15;
        int i = 49 + (z - h * 15);
        f16* srow = sh_sp + (h * 64 + i) * 64;
#pragma unroll
        for (int j = 0; j < 8; ++j) *(h8*)(srow + j * 8) = h8{};
        sh_inv[h * 64 + i] = 0.0f;
    }
    __syncthreads();

    // ---- O = P V, all heads: 32 tiles, 4 per wave (K = 64, 2 WMMA steps) ----
#pragma unroll
    for (int i = 0; i < 4; ++i) {
        int t  = wv * 4 + i;        // p-frag (h,mt) constant over pairs -> CSE
        int h  = t >> 3;
        int mt = (t >> 1) & 3;
        int nt = t & 1;
        v8f c = {};
#pragma unroll
        for (int ks = 0; ks < 2; ++ks) {
            v16h a  = load_frag(sh_sp + (h * 64 + mt * 16) * 64 + ks * 32, 64, lid);
            v16h bf = load_frag(sh_vT + (h * 32 + nt * 16) * 64 + ks * 32, 64, lid);
            c = __builtin_amdgcn_wmma_f32_16x16x32_f16(false, a, false, bf,
                                                       (short)0, c, false, false);
        }
        const float* invp = sh_inv + h * 64 + mt * 16 + hi * 8;
        f16* dst = sh_x + (mt * 16 + hi * 8) * 128 + h * 32 + nt * 16 + ln;
#pragma unroll
        for (int r = 0; r < 8; ++r) dst[r * 128] = (f16)(c[r] * invp[r]);
    }
    __syncthreads();

    // ---- projection: [64x128] x [128x128] + bias, store f32 rows < 49 ----
#pragma unroll
    for (int i = 0; i < 4; ++i) {
        int t = wv + i * 8;
        int nt = t & 7, mt = t >> 3;
        v8f c = {};
#pragma unroll
        for (int ks = 0; ks < 4; ++ks) {
            v16h a  = load_frag(sh_x + mt * 16 * 128 + ks * 32, 128, lid);
            v16h bf = *(const v16h*)(wproj + (size_t)((nt * 4 + ks) * 32 + lid) * 16);
            c = __builtin_amdgcn_wmma_f32_16x16x32_f16(false, a, false, bf,
                                                       (short)0, c, false, false);
        }
        int col = nt * 16 + ln;
        float pb = projb[col];
        float* ob = out + (size_t)b * 49 * 128 + col;
        int row0 = mt * 16 + hi * 8;
#pragma unroll
        for (int r = 0; r < 8; ++r) {
            int row = row0 + r;
            if (row < 49) ob[row * 128] = c[r] + pb;
        }
    }
}

// ---------------------------------------------------------------------------
extern "C" void kernel_launch(void* const* d_in, const int* in_sizes, int n_in,
                              void* d_out, int out_size, void* d_ws, size_t ws_size,
                              hipStream_t stream)
{
    const float* x      = (const float*)d_in[0];
    const float* mask   = (const float*)d_in[1];
    const float* qkv_w  = (const float*)d_in[2];
    const float* qkv_b  = (const float*)d_in[3];
    const float* mlp_w1 = (const float*)d_in[4];
    const float* mlp_b1 = (const float*)d_in[5];
    const float* mlp_w2 = (const float*)d_in[6];
    const float* mlp_b2 = (const float*)d_in[7];
    const float* proj_w = (const float*)d_in[8];
    const float* proj_b = (const float*)d_in[9];

    char* ws = (char*)d_ws;
    f16*   wqkv  = (f16*)(ws + WS_QKVW_OFF);
    f16*   wproj = (f16*)(ws + WS_PROJW_OFF);
    float* bmt   = (float*)(ws + WS_BM_OFF);

    k_swizzle<<<(49152 + 16384 + 255) / 256, 256, 0, stream>>>(qkv_w, proj_w, wqkv, wproj);
    k_biasmask<<<(64 * 2401 + 255) / 256, 256, 0, stream>>>(mask, mlp_w1, mlp_b1,
                                                            mlp_w2, mlp_b2, bmt);
    k_main<<<16384, 256, 0, stream>>>(x, qkv_b, proj_b, wqkv, wproj, bmt, (float*)d_out);
}
